// LSTM_84834194030897
// MI455X (gfx1250) — matrix operand order repaired and checked by reading, not compiled, not run on hardware
//
#include <hip/hip_runtime.h>
#include <math.h>

// LSTM recurrence, T=400000 steps, H=50. Latency-bound serial chain:
// one persistent 128-thread (4-wave) block; W_hh^T held as f16 WMMA
// B-fragments in VGPRs; per-step matvec g = h x W_hh^T done with
// v_wmma_f32_16x16x32_f16 using INLINE-0 as the C operand; the
// x_t*W_ih + (b_ih+b_hh) term is added as one scalar fma+add AFTER the
// WMMA pair (avoids the 7-mov C broadcast and the WMMA WAW hazard nops).
// Activations use the gfx1250 v_tanh_f32 transcendental.

typedef __attribute__((ext_vector_type(16))) _Float16 v16h;
typedef __attribute__((ext_vector_type(8)))  float    v8f;

#define HDIM 50
#define G4   200     // 4*H gate rows
#define NT   13      // ceil(200/16) column tiles

union AFrag { v16h v; uint4 u[2]; };
union BFrag { v16h v; _Float16 e[16]; };

__device__ __forceinline__ float t_tanh(float x) {
#if __has_builtin(__builtin_amdgcn_tanhf)
  return __builtin_amdgcn_tanhf(x);   // v_tanh_f32 (gfx1250 TRANS op)
#else
  return tanhf(x);
#endif
}
__device__ __forceinline__ float sigm(float x) {
  return 0.5f * t_tanh(0.5f * x) + 0.5f;   // sigmoid via tanh unit
}

__global__ __launch_bounds__(128, 1)
void lstm_wmma_kernel(const float* __restrict__ seq,
                      const float* __restrict__ Wih,
                      const float* __restrict__ Whh,
                      const float* __restrict__ bih,
                      const float* __restrict__ bhh,
                      const float* __restrict__ Wlin,
                      const float* __restrict__ blin,
                      float* __restrict__ out, int T)
{
  __shared__ _Float16 hbuf[64];    // h_t (f16), padded 50->64 with zeros
  __shared__ float    gbuf[256];   // gate pre-activations + reduce scratch

  const int tid  = threadIdx.x;
  const int lane = tid & 31;
  // wave id as a true scalar so tile dispatch compiles to s_cbranch
  const int w    = __builtin_amdgcn_readfirstlane(tid >> 5);
  const int hi16 = (lane >> 4) & 1;
  const int n16  = lane & 15;

  // ---- one-time preload: B = Whh^T as f16 fragments (register-resident),
  //      plus per-lane W_ih / (b_ih+b_hh) for the columns this lane covers.
  // Dense 16-bit B layout (32Kx16N): lanes 0-15 -> K 0..7,16..23; lanes 16-31 ->
  // K 8..15,24..31; column n = lane%16; element e <-> K = kbase+e+(e&8).
  // Tiles: k=0..2 -> tile w+4k (always <13); k=3 -> tile 12, wave 0 only.
  BFrag bf[4][2];
  float wi[4], bb[4];
  #pragma unroll
  for (int k = 0; k < 4; ++k) {
    const int tile = w + 4 * k;
    const int n = tile * 16 + n16;         // output gate-row index
    const bool live = (k < 3) || (w == 0);
    wi[k] = 0.f; bb[k] = 0.f;
    if (live && n < G4) { wi[k] = Wih[n]; bb[k] = bih[n] + bhh[n]; }
    #pragma unroll
    for (int c = 0; c < 2; ++c) {
      const int kbase = c * 32 + hi16 * 8;
      #pragma unroll
      for (int e = 0; e < 16; ++e) {
        const int kg = kbase + e + (e & 8);
        float v = 0.f;
        if (live && n < G4 && kg < HDIM) v = Whh[n * HDIM + kg];
        bf[k][c].e[e] = (_Float16)v;
      }
    }
  }

  if (tid < 64) hbuf[tid] = (_Float16)0.f;   // h0 = 0 (+ zero padding)
  float c_st = 0.f;                          // c0 = 0 (per-lane, j = tid)
  float hF   = 0.f;
  __syncthreads();

  for (int t = 0; t < T; ++t) {
    if ((t & 255) == 0) __builtin_prefetch(seq + t + 512, 0, 0);
    const float xt = seq[t];                 // uniform -> s_load, L2 hot

    // A fragments: h broadcast to all 16 rows. Dense 16-bit A layout:
    // lanes 0-15: K 0..7 then 16..23; lanes 16-31: K 8..15 then 24..31.
    AFrag a0, a1;
    {
      const _Float16* hp = hbuf;
      const int b0 = hi16 ? 8 : 0;
      a0.u[0] = *(const uint4*)(hp + b0);
      a0.u[1] = *(const uint4*)(hp + b0 + 16);
      a1.u[0] = *(const uint4*)(hp + b0 + 32);
      a1.u[1] = *(const uint4*)(hp + b0 + 48);
    }

    const v8f zc = {0.f, 0.f, 0.f, 0.f, 0.f, 0.f, 0.f, 0.f};  // inline 0 C

    // Tiles w, w+4, w+8: unconditional (always valid for w in 0..3).
    #pragma unroll
    for (int k = 0; k < 3; ++k) {
      const int tile = w + 4 * k;
      v8f acc = __builtin_amdgcn_wmma_f32_16x16x32_f16(
          false, a0.v, false, bf[k][0].v, (short)0, zc, false, false);
      acc = __builtin_amdgcn_wmma_f32_16x16x32_f16(
          false, a1.v, false, bf[k][1].v, (short)0, acc, false, false);
      // every row of D equals g; lanes l and l+16 write identical values.
      // xg_t + bias added as scalar fma+add after the matrix op.
      gbuf[tile * 16 + n16] = fmaf(xt, wi[k], bb[k]) + acc[0];
    }
    // Tile 12: wave 0 only — scalar branch (w is an SGPR).
    if (w == 0) {
      v8f acc = __builtin_amdgcn_wmma_f32_16x16x32_f16(
          false, a0.v, false, bf[3][0].v, (short)0, zc, false, false);
      acc = __builtin_amdgcn_wmma_f32_16x16x32_f16(
          false, a1.v, false, bf[3][1].v, (short)0, acc, false, false);
      gbuf[12 * 16 + n16] = fmaf(xt, wi[3], bb[3]) + acc[0];
    }
    __syncthreads();

    if (tid < HDIM) {
      const float gi = sigm(gbuf[tid]);
      const float gf = sigm(gbuf[HDIM + tid]);
      const float gg = t_tanh(gbuf[2 * HDIM + tid]);
      const float go = sigm(gbuf[3 * HDIM + tid]);
      c_st = gf * c_st + gi * gg;
      hF   = go * t_tanh(c_st);
      hbuf[tid] = (_Float16)hF;
    }
    __syncthreads();
  }

  // pred = h_{T-1} . W_lin + b_lin  (scalar)
  gbuf[tid] = (tid < HDIM) ? hF * Wlin[tid] : 0.f;
  __syncthreads();
  if (tid == 0) {
    float s = blin[0];
    #pragma unroll
    for (int i = 0; i < 128; ++i) s += gbuf[i];
    out[0] = s;
  }
}

extern "C" void kernel_launch(void* const* d_in, const int* in_sizes, int n_in,
                              void* d_out, int out_size, void* d_ws, size_t ws_size,
                              hipStream_t stream) {
  const float* seq  = (const float*)d_in[0];
  const float* Wih  = (const float*)d_in[1];
  const float* Whh  = (const float*)d_in[2];
  const float* bih  = (const float*)d_in[3];
  const float* bhh  = (const float*)d_in[4];
  const float* Wlin = (const float*)d_in[5];
  const float* blin = (const float*)d_in[6];
  const int T = in_sizes[0];
  lstm_wmma_kernel<<<1, 128, 0, stream>>>(seq, Wih, Whh, bih, bhh, Wlin, blin,
                                          (float*)d_out, T);
}
